// GlobalAbstractionLayer_34763465294634
// MI455X (gfx1250) — matrix operand order repaired and checked by reading, not compiled, not run on hardware
//
#include <hip/hip_runtime.h>

typedef __attribute__((ext_vector_type(16))) __bf16 v16bf;
typedef __attribute__((ext_vector_type(4)))  __bf16 v4bf;
typedef __attribute__((ext_vector_type(8)))  float  v8f;

#define N_POINTS 131072
#define M_TILE   64

__device__ __forceinline__ unsigned short f2bfbits(float f) {
  __bf16 h = (__bf16)f;                 // hardware cvt (RNE)
  unsigned short s;
  __builtin_memcpy(&s, &h, 2);
  return s;
}

// Pack W (K x N row-major f32, row stride rs) into WMMA B-fragment order:
// per 32x16 tile (kt,nt), lane l holds column nt*16+(l&15),
// K = kt*32 + (l>>4)*8 + e for e=0..7, kt*32 + 16 + (l>>4)*8 + (e-8) for e=8..15,
// stored contiguously (32 bytes/lane).
__global__ void pack_w_kernel(const float* __restrict__ W, int K, int N, int rs,
                              unsigned short* __restrict__ dst) {
  int ntn = N >> 4;
  int tid = blockIdx.x * blockDim.x + threadIdx.x;
  int total = (K >> 5) * ntn * 32;
  if (tid >= total) return;
  int lane = tid & 31;
  int t = tid >> 5;
  int nt = t % ntn, kt = t / ntn;
  int col = nt * 16 + (lane & 15);
  int kb = kt * 32 + ((lane >> 4) << 3);
  unsigned short* o = dst + (size_t)tid * 16;
#pragma unroll
  for (int e = 0; e < 16; ++e)
    o[e] = f2bfbits(W[(size_t)(kb + e + (e >= 8 ? 8 : 0)) * rs + col]);
}

__global__ void zero_out_kernel(float* __restrict__ out, int n) {
  int i = blockIdx.x * blockDim.x + threadIdx.x;
  if (i < n) out[i] = 0.0f;
}

// A fragment: lane l -> row mt*16+(l&15); elems 0..7 = K kb..kb+7, 8..15 = kb+16..kb+23,
// kb = kt*32 + (l>>4)*8   (ISA 7.12.2, 16-bit A 16x32 layout)
#define LOAD_A(dstv, base, ld, mt, kt)                                         \
  {                                                                            \
    const unsigned short* _p = (base) + ((mt) * 16 + (lane & 15)) * (ld) +     \
                               (kt) * 32 + ((lane >> 4) << 3);                 \
    union { v16bf v; uint4 q[2]; } _u;                                         \
    _u.q[0] = *(const uint4*)(_p);                                             \
    _u.q[1] = *(const uint4*)(_p + 16);                                        \
    dstv = _u.v;                                                               \
  }

// B fragment: contiguous 32 bytes per lane from the packed weights
#define LOAD_B(dstv, wpk, ntn, kt, nt)                                         \
  {                                                                            \
    const unsigned short* _p =                                                 \
        (wpk) + ((size_t)(((kt) * (ntn) + (nt)) * 32 + lane) << 4);            \
    union { v16bf v; uint4 q[2]; } _u;                                         \
    _u.q[0] = *(const uint4*)(_p);                                             \
    _u.q[1] = *(const uint4*)(_p + 16);                                        \
    dstv = _u.v;                                                               \
  }

__global__ __launch_bounds__(256)
void pointnet_fused_kernel(const float* __restrict__ x,
                           const float* __restrict__ pos,
                           const int*   __restrict__ batch,
                           const float* __restrict__ W1,   // rows 256..258 used
                           const float* __restrict__ b1,
                           const float* __restrict__ b2,
                           const float* __restrict__ b3,
                           const unsigned short* __restrict__ w1pk,
                           const unsigned short* __restrict__ w2pk,
                           const unsigned short* __restrict__ w3pk,
                           float* __restrict__ out) {
  extern __shared__ unsigned char smem[];
  unsigned short* bufA = (unsigned short*)smem;                  // 64*512 bf16 (64KB)
  unsigned short* bufB = (unsigned short*)(smem + 65536);        // 64*256 bf16 (32KB)
  float* posS = (float*)(smem + 65536 + 32768);                  // 64*3 f32
  float* w1bS = (float*)(smem + 65536 + 32768 + 768);            // 3*256 f32

  const int tid  = threadIdx.x;
  const int lane = tid & 31;
  const int wave = tid >> 5;
  const int row0 = blockIdx.x * M_TILE;

  // ---- stage x tile as bf16 (vectorized: b128 load -> packed cvt -> b64 store)
  {
    const float4* x4 = (const float4*)(x + (size_t)row0 * 256);
    for (int i = tid; i < M_TILE * 64; i += 256) {
      float4 f = x4[i];
      v4bf h;
      h[0] = (__bf16)f.x; h[1] = (__bf16)f.y;
      h[2] = (__bf16)f.z; h[3] = (__bf16)f.w;
      *(v4bf*)(bufA + (size_t)i * 4) = h;
    }
  }
  for (int i = tid; i < M_TILE * 3; i += 256)
    posS[i] = pos[(size_t)row0 * 3 + i];
  for (int i = tid; i < 3 * 256; i += 256)
    w1bS[i] = W1[(size_t)(256 + (i >> 8)) * 256 + (i & 255)];
  __syncthreads();

  const int hi8 = (lane >> 4) << 3;   // 0 or 8: row-half offset in C/D layout

  // ---- Layer 1: h1 = relu(x @ W1[0:256] + pos @ W1[256:259] + b1)  [64,256]
  // Wave owns 2 N-tiles: hoist the 8 B-fragments, loop all 4 M-tiles.
  for (int i = 0; i < 2; ++i) {
    const int nt = wave * 2 + i;
    const int col = nt * 16 + (lane & 15);
    v16bf bfr[8];
#pragma unroll
    for (int kt = 0; kt < 8; ++kt) LOAD_B(bfr[kt], w1pk, 16, kt, nt);
    const float bb = b1[col];
    const float wb0 = w1bS[col], wb1 = w1bS[256 + col], wb2 = w1bS[512 + col];
    for (int mt = 0; mt < 4; ++mt) {
      const int rbase = mt * 16 + hi8;
      v8f acc;
#pragma unroll
      for (int r = 0; r < 8; ++r) {
        int row = rbase + r;
        acc[r] = bb + posS[row * 3 + 0] * wb0 + posS[row * 3 + 1] * wb1 +
                 posS[row * 3 + 2] * wb2;
      }
#pragma unroll
      for (int kt = 0; kt < 8; ++kt) {
        v16bf a;
        LOAD_A(a, bufA, 256, mt, kt);
        acc = __builtin_amdgcn_wmma_f32_16x16x32_bf16(
            false, a, false, bfr[kt], (short)0, acc, false, false);
      }
#pragma unroll
      for (int r = 0; r < 8; ++r) {
        __bf16 h = (__bf16)fmaxf(acc[r], 0.0f);
        ((__bf16*)bufB)[(rbase + r) * 256 + col] = h;
      }
    }
  }
  __syncthreads();

  // ---- Layer 2: h2 = relu(h1 @ W2 + b2)  [64,512] (overwrites x region)
  // Wave owns 4 N-tiles: hoist the 8 B-fragments, loop all 4 M-tiles.
  for (int i = 0; i < 4; ++i) {
    const int nt = wave * 4 + i;
    const int col = nt * 16 + (lane & 15);
    v16bf bfr[8];
#pragma unroll
    for (int kt = 0; kt < 8; ++kt) LOAD_B(bfr[kt], w2pk, 32, kt, nt);
    const float bb = b2[col];
    for (int mt = 0; mt < 4; ++mt) {
      const int rbase = mt * 16 + hi8;
      v8f acc;
#pragma unroll
      for (int r = 0; r < 8; ++r) acc[r] = bb;
#pragma unroll
      for (int kt = 0; kt < 8; ++kt) {
        v16bf a;
        LOAD_A(a, bufB, 256, mt, kt);
        acc = __builtin_amdgcn_wmma_f32_16x16x32_bf16(
            false, a, false, bfr[kt], (short)0, acc, false, false);
      }
#pragma unroll
      for (int r = 0; r < 8; ++r) {
        __bf16 h = (__bf16)fmaxf(acc[r], 0.0f);
        ((__bf16*)bufA)[(rbase + r) * 512 + col] = h;
      }
    }
  }
  __syncthreads();

  // ---- Layer 3: h3 = relu(h2 @ W3 + b3) fused with segment max  [*,1024]
  // Wave owns 8 N-tiles: hoist the 16 B-fragments (128 VGPRs), loop all 4
  // M-tiles reading A from LDS; fold ReLU + row-max into the epilogue.
  {
    const int seg = batch[row0];
    for (int i = 0; i < 8; ++i) {
      const int nt = wave * 8 + i;
      const int col = nt * 16 + (lane & 15);
      v16bf bfr[16];
#pragma unroll
      for (int kt = 0; kt < 16; ++kt) LOAD_B(bfr[kt], w3pk, 64, kt, nt);
      const float bb = b3[col];
      float cmax = 0.0f;                     // >=0 => ReLU folded into max
      for (int mt = 0; mt < 4; ++mt) {
        v8f acc;
#pragma unroll
        for (int r = 0; r < 8; ++r) acc[r] = bb;
#pragma unroll
        for (int kt = 0; kt < 16; ++kt) {
          v16bf a;
          LOAD_A(a, bufA, 512, mt, kt);
          acc = __builtin_amdgcn_wmma_f32_16x16x32_bf16(
              false, a, false, bfr[kt], (short)0, acc, false, false);
        }
#pragma unroll
        for (int r = 0; r < 8; ++r) cmax = fmaxf(cmax, acc[r]);
      }
      // combine the two row-halves (lane l and l+16 share the same column)
      float o = __shfl_xor(cmax, 16, 32);
      float m = fmaxf(cmax, o);
      if (lane < 16) {
        // all values >= 0 -> float order == unsigned-bit order
        atomicMax((unsigned int*)(out + (size_t)seg * 1024 + col),
                  __float_as_uint(m));
      }
    }
  }
}

extern "C" void kernel_launch(void* const* d_in, const int* in_sizes, int n_in,
                              void* d_out, int out_size, void* d_ws, size_t ws_size,
                              hipStream_t stream) {
  const float* x     = (const float*)d_in[0];
  const float* pos   = (const float*)d_in[1];
  const int*   batch = (const int*)d_in[2];
  const float* W1    = (const float*)d_in[3];
  const float* b1    = (const float*)d_in[4];
  const float* W2    = (const float*)d_in[5];
  const float* b2    = (const float*)d_in[6];
  const float* W3    = (const float*)d_in[7];
  const float* b3    = (const float*)d_in[8];
  float* out = (float*)d_out;

  unsigned char* ws = (unsigned char*)d_ws;
  unsigned short* w1pk = (unsigned short*)ws;                      // 256*256*2 = 128KB
  unsigned short* w2pk = (unsigned short*)(ws + 131072);           // 256*512*2 = 256KB
  unsigned short* w3pk = (unsigned short*)(ws + 131072 + 262144);  // 512*1024*2 = 1MB

  zero_out_kernel<<<(32 * 1024 + 255) / 256, 256, 0, stream>>>(out, 32 * 1024);

  int tot1 = (256 / 32) * (256 / 16) * 32;
  pack_w_kernel<<<(tot1 + 255) / 256, 256, 0, stream>>>(W1, 256, 256, 256, w1pk);
  int tot2 = (256 / 32) * (512 / 16) * 32;
  pack_w_kernel<<<(tot2 + 255) / 256, 256, 0, stream>>>(W2, 256, 512, 512, w2pk);
  int tot3 = (512 / 32) * (1024 / 16) * 32;
  pack_w_kernel<<<(tot3 + 255) / 256, 256, 0, stream>>>(W3, 512, 1024, 1024, w3pk);

  size_t shmem = 65536 + 32768 + 768 + 3072;  // ~100KB of the 320KB WGP LDS
  pointnet_fused_kernel<<<N_POINTS / M_TILE, 256, shmem, stream>>>(
      x, pos, batch, W1, b1, b2, b3, w1pk, w2pk, w3pk, out);
}
